// GAT_36713380446370
// MI455X (gfx1250) — compile-verified, matrix operand order checked by reading, not run on hardware
//
#include <hip/hip_runtime.h>

#define NNODES 4096
#define FDIM   512
#define DDIM   64
#define NHEADS 8

typedef __attribute__((ext_vector_type(16))) __bf16       v16bf;
typedef __attribute__((ext_vector_type(8)))  float        v8f;
typedef __attribute__((ext_vector_type(4)))  unsigned int u32x4;
typedef __attribute__((ext_vector_type(4)))  float        f32x4;
typedef __attribute__((ext_vector_type(4)))  int          i32x4;

union Frag16 { v16bf v; u32x4 q[2]; };

__device__ __forceinline__ unsigned short f2bfu(float f) {
  unsigned int u = __builtin_bit_cast(unsigned int, f);
  u = (u + 0x7FFFu + ((u >> 16) & 1u)) >> 16;
  return (unsigned short)u;
}
__device__ __forceinline__ __bf16 f2bf(float f) {
  return __builtin_bit_cast(__bf16, f2bfu(f));
}
__device__ __forceinline__ float bf2f(unsigned short u) {
  unsigned int x = ((unsigned int)u) << 16;
  return __builtin_bit_cast(float, x);
}

// ---------------------------------------------------------------------------
// fp32 -> bf16 elementwise convert
__global__ void cvt_f32_bf16_kernel(const float* __restrict__ src,
                                    unsigned short* __restrict__ dst, int n) {
  int i = blockIdx.x * blockDim.x + threadIdx.x;
  if (i < n) dst[i] = f2bfu(src[i]);
}

// W [8][512][64] fp32 -> Wt [8][64][512] bf16 (K contiguous for B-fragments)
__global__ void wt_transpose_kernel(const float* __restrict__ W,
                                    unsigned short* __restrict__ wt) {
  int i = blockIdx.x * blockDim.x + threadIdx.x;   // 8*512*64
  int h = i >> 15;
  int rem = i & 32767;
  int f = rem >> 6;
  int d = rem & 63;
  wt[((long)h * DDIM + d) * FDIM + f] = f2bfu(W[i]);
}

// ---------------------------------------------------------------------------
// Projection: Ht[head][d][n] (bf16, transposed) = (x @ W[head])^T
// one wave per (16-row tile, head); K loop over FDIM in steps of 32
__global__ void __launch_bounds__(32)
gat_proj_kernel(const unsigned short* __restrict__ xb,
                const unsigned short* __restrict__ wt,
                unsigned short* __restrict__ ht) {
  const int lane = threadIdx.x & 31;
  const int hl   = lane >> 4;     // which half of the wave
  const int mrow = lane & 15;     // M (or N) index within 16
  const int n0   = blockIdx.x * 16;
  const int head = blockIdx.y;

  v8f acc[4] = {};
  const unsigned short* wh = wt + (long)head * DDIM * FDIM;

  for (int kf = 0; kf < FDIM; kf += 32) {
    // A fragment: x rows n0..n0+15, K=kf..kf+31 (16-bit A layout)
    const unsigned short* ap = xb + (long)(n0 + mrow) * FDIM + kf + hl * 8;
    Frag16 af;
    af.q[0] = *(const u32x4*)(ap);
    af.q[1] = *(const u32x4*)(ap + 16);
#pragma unroll
    for (int g = 0; g < 4; ++g) {
      // B fragment: Wt row d = g*16+mrow, K contiguous
      const unsigned short* bp = wh + (long)(g * 16 + mrow) * FDIM + kf + hl * 8;
      Frag16 bfr;
      bfr.q[0] = *(const u32x4*)(bp);
      bfr.q[1] = *(const u32x4*)(bp + 16);
      acc[g] = __builtin_amdgcn_wmma_f32_16x16x32_bf16(
          false, af.v, false, bfr.v, (short)0, acc[g], false, false);
    }
  }

  // C layout: vgpr r -> row M = r + 8*hl, col N = g*16 + mrow; store transposed
  unsigned short* hh = ht + (long)head * DDIM * NNODES;
#pragma unroll
  for (int r = 0; r < 8; ++r) {
    int n = n0 + hl * 8 + r;
#pragma unroll
    for (int g = 0; g < 4; ++g)
      hh[(long)(g * 16 + mrow) * NNODES + n] = f2bfu(acc[g][r]);
  }
}

// ---------------------------------------------------------------------------
// f1[h][n] = Ht[h][:,n] . a1[h], f2 likewise
__global__ void f1f2_kernel(const unsigned short* __restrict__ ht,
                            const float* __restrict__ a1,
                            const float* __restrict__ a2,
                            float* __restrict__ f1, float* __restrict__ f2) {
  int i = blockIdx.x * blockDim.x + threadIdx.x;   // 8*4096
  int head = i >> 12, n = i & 4095;
  const unsigned short* hh = ht + (long)head * DDIM * NNODES + n;
  float s1 = 0.f, s2 = 0.f;
#pragma unroll 8
  for (int d = 0; d < DDIM; ++d) {
    float hv = bf2f(hh[(long)d * NNODES]);
    s1 += hv * a1[head * DDIM + d];
    s2 += hv * a2[head * DDIM + d];
  }
  f1[i] = s1;
  f2[i] = s2;
}

// ---------------------------------------------------------------------------
// Fused masked-softmax aggregation (flash style):
//   out[i,:] = elu( softmax_j( mask(leakyrelu(f1_i+f2_j)) ) @ H )
// block = 256 threads = 8 waves; wave w handles head w, rows i0..i0+31
// adj 32x32 tile staged once in LDS, shared by all 8 heads.
__global__ void __launch_bounds__(256)
gat_attn_kernel(const float* __restrict__ f1g, const float* __restrict__ f2g,
                const int* __restrict__ adj, const unsigned short* __restrict__ ht,
                unsigned short* __restrict__ xout, float* __restrict__ gout,
                int mode) {
  __shared__ __align__(16) int adjT[32 * 32];

  const int tid  = threadIdx.x;
  const int head = tid >> 5;
  const int lane = tid & 31;
  const int hl   = lane >> 4;
  const int mrow = lane & 15;
  const int i0   = blockIdx.x * 32;

  const float f1t0 = f1g[head * NNODES + i0 + mrow];
  const float f1t1 = f1g[head * NNODES + i0 + 16 + mrow];
  const float* f2h = f2g + head * NNODES;
  const unsigned short* hb = ht + (long)head * DDIM * NNODES;

  float mt[2] = {-3.0e38f, -3.0e38f};
  float lt[2] = {0.f, 0.f};
  v8f acc[2][4] = {};

  // cooperative adj staging indices: 32 rows x 32 cols ints, int4 per thread
  const int lr = tid >> 3;
  const int lc = (tid & 7) * 4;
  const long arowbase = (long)(i0 + lr) * NNODES + lc;

  for (int j0 = 0; j0 < NNODES; j0 += 32) {
    __syncthreads();
    *(i32x4*)&adjT[lr * 32 + lc] = *(const i32x4*)(adj + arowbase + j0);
    if (j0 + 32 < NNODES) {
      // gfx1250 global_prefetch: pull next adj tile and next Ht chunk toward L0
      __builtin_prefetch((const void*)(adj + arowbase + j0 + 32), 0, 1);
      __builtin_prefetch((const void*)(hb + (long)mrow * NNODES + j0 + 32 + hl * 8), 0, 1);
    }
    __syncthreads();

    const int jb = j0 + hl * 8;
    f32x4 fa = *(const f32x4*)(f2h + jb);
    f32x4 fb = *(const f32x4*)(f2h + jb + 4);
    f32x4 fc = *(const f32x4*)(f2h + jb + 16);
    f32x4 fd = *(const f32x4*)(f2h + jb + 20);
    float fvv[16];
#pragma unroll
    for (int e = 0; e < 4; ++e) {
      fvv[e] = fa[e]; fvv[e + 4] = fb[e]; fvv[e + 8] = fc[e]; fvv[e + 12] = fd[e];
    }

    // B fragments from transposed H (shared by both row-tiles)
    Frag16 bfr[4];
#pragma unroll
    for (int g = 0; g < 4; ++g) {
      const unsigned short* bp = hb + (long)(g * 16 + mrow) * NNODES + jb;
      bfr[g].q[0] = *(const u32x4*)(bp);
      bfr[g].q[1] = *(const u32x4*)(bp + 16);
    }

#pragma unroll
    for (int t = 0; t < 2; ++t) {
      const int* arow = &adjT[(t * 16 + mrow) * 32 + hl * 8];
      i32x4 aA = *(const i32x4*)(arow);
      i32x4 aB = *(const i32x4*)(arow + 4);
      i32x4 aC = *(const i32x4*)(arow + 16);
      i32x4 aD = *(const i32x4*)(arow + 20);
      int avv[16];
#pragma unroll
      for (int e = 0; e < 4; ++e) {
        avv[e] = aA[e]; avv[e + 4] = aB[e]; avv[e + 8] = aC[e]; avv[e + 12] = aD[e];
      }

      const float f1i = (t == 0) ? f1t0 : f1t1;
      float s[16];
#pragma unroll
      for (int e = 0; e < 16; ++e) {
        float v = f1i + fvv[e];
        v = (v >= 0.f) ? v : 0.2f * v;          // LeakyReLU(0.2)
        s[e] = (avv[e] > 0) ? v : -9.0e15f;     // NEG_BIG mask (matches ref)
      }

      float tmax = s[0];
#pragma unroll
      for (int e = 1; e < 16; ++e) tmax = fmaxf(tmax, s[e]);
      tmax = fmaxf(tmax, __shfl_xor(tmax, 16, 32));
      float mnew  = fmaxf(mt[t], tmax);
      float alpha = __expf(mt[t] - mnew);

      Frag16 pf;
      float psum = 0.f;
#pragma unroll
      for (int e = 0; e < 16; ++e) {
        float p = __expf(s[e] - mnew);
        psum += p;
        pf.v[e] = f2bf(p);
      }
      psum += __shfl_xor(psum, 16, 32);
      lt[t] = lt[t] * alpha + psum;
      mt[t] = mnew;

      // rescale accumulators: row M = r + 8*hl needs alpha from lane (r+8*hl)
#pragma unroll
      for (int r = 0; r < 8; ++r) {
        float ar = __shfl(alpha, hl * 8 + r, 32);
        acc[t][0][r] *= ar; acc[t][1][r] *= ar;
        acc[t][2][r] *= ar; acc[t][3][r] *= ar;
      }
#pragma unroll
      for (int g = 0; g < 4; ++g)
        acc[t][g] = __builtin_amdgcn_wmma_f32_16x16x32_bf16(
            false, pf.v, false, bfr[g].v, (short)0, acc[t][g], false, false);
    }
  }

  // epilogue: normalize, elu, write
#pragma unroll
  for (int t = 0; t < 2; ++t) {
    float invl = 1.f / lt[t];
#pragma unroll
    for (int r = 0; r < 8; ++r) {
      float ir = __shfl(invl, hl * 8 + r, 32);
      int n = i0 + t * 16 + hl * 8 + r;
#pragma unroll
      for (int g = 0; g < 4; ++g) {
        float v = acc[t][g][r] * ir;
        v = (v > 0.f) ? v : (__expf(v) - 1.f);  // elu
        int d = g * 16 + mrow;
        if (mode == 0)
          xout[(long)n * FDIM + head * DDIM + d] = f2bfu(v);   // concat heads
        else
          gout[((long)head * NNODES + n) * DDIM + d] = v;      // per-head fp32
      }
    }
  }
}

// ---------------------------------------------------------------------------
// out = elu(mean over heads)
__global__ void mean_elu_kernel(const float* __restrict__ g,
                                float* __restrict__ out) {
  int i = blockIdx.x * blockDim.x + threadIdx.x;   // 4096*64
  int n = i >> 6, c = i & 63;
  float s = 0.f;
#pragma unroll
  for (int h = 0; h < NHEADS; ++h) s += g[((long)h * NNODES + n) * DDIM + c];
  float v = s * (1.0f / NHEADS);
  out[i] = (v > 0.f) ? v : (__expf(v) - 1.f);
}

// ---------------------------------------------------------------------------
extern "C" void kernel_launch(void* const* d_in, const int* in_sizes, int n_in,
                              void* d_out, int out_size, void* d_ws, size_t ws_size,
                              hipStream_t stream) {
  const float* x   = (const float*)d_in[0];
  const int*   adj = (const int*)d_in[1];
  const float* W1  = (const float*)d_in[2];
  const float* a11 = (const float*)d_in[3];
  const float* a21 = (const float*)d_in[4];
  const float* W2  = (const float*)d_in[5];
  const float* a12 = (const float*)d_in[6];
  const float* a22 = (const float*)d_in[7];
  const float* W3  = (const float*)d_in[8];
  const float* a13 = (const float*)d_in[9];
  const float* a23 = (const float*)d_in[10];

  char* ws = (char*)d_ws;
  unsigned short* xb0 = (unsigned short*)(ws);                 // 4 MB
  unsigned short* xb1 = (unsigned short*)(ws + 4194304);       // 4 MB
  unsigned short* wt  = (unsigned short*)(ws + 8388608);       // 512 KB
  unsigned short* ht  = (unsigned short*)(ws + 8912896);       // 4 MB
  float*          f1  = (float*)(ws + 13107200);               // 128 KB
  float*          f2  = (float*)(ws + 13238272);               // 128 KB
  float*          gb  = (float*)(ws + 13369344);               // 8 MB

  const dim3 projGrid(NNODES / 16, NHEADS, 1), projBlk(32, 1, 1);
  const dim3 attnGrid(NNODES / 32, 1, 1),      attnBlk(256, 1, 1);

  cvt_f32_bf16_kernel<<<(NNODES * FDIM) / 256, 256, 0, stream>>>(x, xb0, NNODES * FDIM);

  // ---- layer 1 ----
  wt_transpose_kernel<<<(NHEADS * FDIM * DDIM) / 256, 256, 0, stream>>>(W1, wt);
  gat_proj_kernel<<<projGrid, projBlk, 0, stream>>>(xb0, wt, ht);
  f1f2_kernel<<<(NHEADS * NNODES) / 256, 256, 0, stream>>>(ht, a11, a21, f1, f2);
  gat_attn_kernel<<<attnGrid, attnBlk, 0, stream>>>(f1, f2, adj, ht, xb1, nullptr, 0);

  // ---- layer 2 ----
  wt_transpose_kernel<<<(NHEADS * FDIM * DDIM) / 256, 256, 0, stream>>>(W2, wt);
  gat_proj_kernel<<<projGrid, projBlk, 0, stream>>>(xb1, wt, ht);
  f1f2_kernel<<<(NHEADS * NNODES) / 256, 256, 0, stream>>>(ht, a12, a22, f1, f2);
  gat_attn_kernel<<<attnGrid, attnBlk, 0, stream>>>(f1, f2, adj, ht, xb0, nullptr, 0);

  // ---- layer 3 (per-head elu'd outputs to gb, then mean+elu) ----
  wt_transpose_kernel<<<(NHEADS * FDIM * DDIM) / 256, 256, 0, stream>>>(W3, wt);
  gat_proj_kernel<<<projGrid, projBlk, 0, stream>>>(xb0, wt, ht);
  f1f2_kernel<<<(NHEADS * NNODES) / 256, 256, 0, stream>>>(ht, a13, a23, f1, f2);
  gat_attn_kernel<<<attnGrid, attnBlk, 0, stream>>>(f1, f2, adj, ht, nullptr, gb, 1);
  mean_elu_kernel<<<(NNODES * DDIM) / 256, 256, 0, stream>>>(gb, (float*)d_out);
}
